// DGNN_996432413629
// MI455X (gfx1250) — compile-verified, matrix operand order checked by reading, not run on hardware
//
#include <hip/hip_runtime.h>
#include <hip/hip_bf16.h>

typedef __attribute__((ext_vector_type(16))) __bf16 v16bf;
typedef __attribute__((ext_vector_type(8)))  float  v8f;

#define N_NODES 200000
#define M_DIM   200
#define DM_DIM  100
#define DE_DIM  172
#define E_EDGES 8192
#define K_NBR   10
#define G3M     600
#define TAUF    2.0f

// ---------- WMMA fragment index maps (CDNA5 16-bit A 16x32 layout) ----------
// VGPR v (j=2v+p elements), lanes 0-15: K = 2v+p (v<4) else 16+2(v-4)+p ; lanes 16-31: +8
__device__ __forceinline__ int kmap(int j, int hv) {
  int v = j >> 1, p = j & 1;
  int base = (v < 4) ? (2 * v) : (16 + 2 * (v - 4));
  return base + 8 * hv + p;
}
// inverse: k-offset within 32 -> (half, element j)
__device__ __forceinline__ void kinv(int ko, int& hv, int& j) {
  hv = (ko >> 3) & 1;
  j  = (ko & 7) + ((ko & 16) ? 8 : 0);
}

__device__ __forceinline__ v8f wmma_bf16(v16bf a, v16bf b, v8f c) {
  return __builtin_amdgcn_wmma_f32_16x16x32_bf16(false, a, false, b, (short)0, c, false, false);
}

__device__ __forceinline__ float sigmoidf_(float x) { return 1.0f / (1.0f + expf(-x)); }

// ---------- weight packing: f32 [Krows x Ncols] -> bf16 fragment-contiguous ----------
// layout: ((kt*nCt + ct)*32 + lane)*16 + j ; value = W[kt*32 + kmap(j,lane>>4)][ct*16 + (lane&15)]
__global__ void pack_weight_kernel(const float* __restrict__ W, __bf16* __restrict__ out,
                                   int Krows, int Ncols, int nKt, int nCt) {
  int tid = blockIdx.x * blockDim.x + threadIdx.x;
  int total = nKt * nCt * 512;
  if (tid >= total) return;
  int j    = tid & 15;
  int lane = (tid >> 4) & 31;
  int ct   = (tid >> 9) % nCt;
  int kt   = (tid >> 9) / nCt;
  int k = kt * 32 + kmap(j, lane >> 4);
  int n = ct * 16 + (lane & 15);
  float v = (k < Krows && n < Ncols) ? W[(size_t)k * Ncols + n] : 0.0f;
  out[tid] = (__bf16)v;
}

// ---------- last-occurrence machinery ----------
__global__ void init_lastocc_kernel(int* lo, int n) {
  int t = blockIdx.x * blockDim.x + threadIdx.x;
  if (t < n) lo[t] = -1;
}
__global__ void scatter_last_kernel(const int* __restrict__ src, const int* __restrict__ dst,
                                    int* lo_s, int* lo_g) {
  int e = blockIdx.x * blockDim.x + threadIdx.x;
  if (e >= E_EDGES) return;
  atomicMax(&lo_s[src[e]], e);
  atomicMax(&lo_g[dst[e]], e);
}
__global__ void is_last_kernel(const int* __restrict__ src, const int* __restrict__ dst,
                               const int* __restrict__ lo_s, const int* __restrict__ lo_g,
                               float* il_s, float* il_g) {
  int e = blockIdx.x * blockDim.x + threadIdx.x;
  if (e >= E_EDGES) return;
  il_s[e] = (lo_s[src[e]] == e) ? 1.0f : 0.0f;
  il_g[e] = (lo_g[dst[e]] == e) ? 1.0f : 0.0f;
}

// ---------- shared WMMA core: LDS-staged A tile x fragment-packed B ----------
template <int NKT, int NCT>
__device__ __forceinline__ void wmma_core_store(const __bf16* lA,
                                                const __bf16* __restrict__ Bsw,
                                                const float* __restrict__ bias,
                                                float* __restrict__ out,
                                                int rowBase, int Ncols, int act) {
  const int wave = threadIdx.x >> 5, lane = threadIdx.x & 31;
  const int ct = blockIdx.y * 4 + wave;
  if (ct >= NCT) return;
  const __bf16* aBase = lA + lane * 16;                        // + kt*512  (imm offsets)
  const __bf16* bBase = Bsw + ((size_t)ct * 32 + lane) * 16;   // + kt*NCT*512 (imm offsets)
  v8f c = {};
#pragma unroll
  for (int kt = 0; kt < NKT; ++kt) {
    v16bf a = *(const v16bf*)(aBase + kt * 512);
    v16bf b = *(const v16bf*)(bBase + (size_t)kt * NCT * 512);
    c = wmma_bf16(a, b, c);
  }
  const int hv = lane >> 4, col = ct * 16 + (lane & 15);
  if (col >= Ncols) return;
  const float bv = bias ? bias[col] : 0.0f;
#pragma unroll
  for (int r = 0; r < 8; ++r) {
    float x = c[r] + bv;
    if (act == 1) x = tanhf(x);
    else if (act == 2) x = fmaxf(x, 0.0f);
    out[(size_t)(rowBase + r + 8 * hv) * Ncols + col] = x;
  }
}

// ---------- merge GEMM: tanh(concat(memA[idx],memB[idx]) @ W + b) -> [E x 200] ----------
// K = 400 (pad 416, 13 k-tiles), N = 200 (13 col tiles). 4 waves/block, 1 col tile/wave.
__global__ void merge_gemm_kernel(const float* __restrict__ memA, const float* __restrict__ memB,
                                  const int* __restrict__ nodes,
                                  const __bf16* __restrict__ Wsw, const float* __restrict__ bias,
                                  float* __restrict__ out) {
  __shared__ __bf16 lA[13 * 512];
  const int rowBase = blockIdx.x * 16;
  const int Ktot = 416;
  for (int t = threadIdx.x; t < 16 * Ktot; t += blockDim.x) {
    int row = t / Ktot, k = t - row * Ktot;
    int node = nodes[rowBase + row];
    float val = 0.0f;
    if (k < M_DIM)          val = memA[(size_t)node * M_DIM + k];
    else if (k < 2 * M_DIM) val = memB[(size_t)node * M_DIM + (k - M_DIM)];
    int hv, j; kinv(k & 31, hv, j);
    lA[((k >> 5) * 32 + ((hv << 4) | row)) * 16 + j] = (__bf16)val;
  }
  __syncthreads();
  wmma_core_store<13, 13>(lA, Wsw, bias, out, rowBase, M_DIM, /*tanh*/1);
}

// ---------- message GEMM: relu(concat(bufA,bufB,ef[eidx]) @ Wmsg + b) -> [E x 100] ----------
// K = 572 (pad 576, 18 k-tiles), N = 100 (7 col tiles)
__global__ void msg_gemm_kernel(const float* __restrict__ bufA, const float* __restrict__ bufB,
                                const float* __restrict__ edge_feats, const int* __restrict__ eidx,
                                const __bf16* __restrict__ Wsw, const float* __restrict__ bias,
                                float* __restrict__ out) {
  __shared__ __bf16 lA[18 * 512];
  const int rowBase = blockIdx.x * 16;
  const int Ktot = 576;
  for (int t = threadIdx.x; t < 16 * Ktot; t += blockDim.x) {
    int row = t / Ktot, k = t - row * Ktot;
    int e = rowBase + row;
    float val = 0.0f;
    if (k < 200)      val = bufA[(size_t)e * M_DIM + k];
    else if (k < 400) val = bufB[(size_t)e * M_DIM + (k - 200)];
    else if (k < 572) val = edge_feats[(size_t)eidx[e] * DE_DIM + (k - 400)];
    int hv, j; kinv(k & 31, hv, j);
    lA[((k >> 5) * 32 + ((hv << 4) | row)) * 16 + j] = (__bf16)val;
  }
  __syncthreads();
  wmma_core_store<18, 7>(lA, Wsw, bias, out, rowBase, DM_DIM, /*relu*/2);
}

// ---------- generic dense GEMM: out = act(A[idx?] @ Bsw (+ bias)) ----------
// act: 0=none, 1=tanh
template <int NKT, int NCT>
__global__ void dense_gemm_kernel(const float* __restrict__ A, const int* __restrict__ idx,
                                  int Kdim, int lda,
                                  const __bf16* __restrict__ Bsw,
                                  const float* __restrict__ bias,
                                  float* __restrict__ out, int Ncols, int act) {
  __shared__ __bf16 lA[NKT * 512];
  const int rowBase = blockIdx.x * 16;
  const int Ktot = NKT * 32;
  for (int t = threadIdx.x; t < 16 * Ktot; t += blockDim.x) {
    int row = t / Ktot, k = t - row * Ktot;
    int gr = rowBase + row;
    int srow = idx ? idx[gr] : gr;
    float val = (k < Kdim) ? A[(size_t)srow * lda + k] : 0.0f;
    int hv, j; kinv(k & 31, hv, j);
    lA[((k >> 5) * 32 + ((hv << 4) | row)) * 16 + j] = (__bf16)val;
  }
  __syncthreads();
  wmma_core_store<NKT, NCT>(lA, Bsw, bias, out, rowBase, Ncols, act);
}

// ---------- GRU elementwise + delta scatter ----------
__global__ void gru_scatter_kernel(const float* __restrict__ gx, const float* __restrict__ gh,
                                   const float* __restrict__ mem, const int* __restrict__ nodes,
                                   const float* __restrict__ is_last, float* mem2) {
  int tid = blockIdx.x * blockDim.x + threadIdx.x;
  if (tid >= E_EDGES * M_DIM) return;
  int e = tid / M_DIM, m = tid - e * M_DIM;
  float il = is_last[e];
  if (il == 0.0f) return;
  int node = nodes[e];
  size_t gb = (size_t)e * G3M;
  float r = sigmoidf_(gx[gb + m] + gh[gb + m]);
  float z = sigmoidf_(gx[gb + M_DIM + m] + gh[gb + M_DIM + m]);
  float n = tanhf(gx[gb + 2 * M_DIM + m] + r * gh[gb + 2 * M_DIM + m]);
  float h = mem[(size_t)node * M_DIM + m];
  float nh = (1.0f - z) * n + z * h;
  unsafeAtomicAdd(&mem2[(size_t)node * M_DIM + m], il * (nh - h));
}

// ---------- neighbor propagation scatter ----------
__global__ void prop_scatter_kernel(const float* __restrict__ prop, const int* __restrict__ nodes,
                                    const int* __restrict__ neighbors,
                                    const float* __restrict__ times,
                                    const float* __restrict__ last_upd,
                                    const float* __restrict__ is_last, float* mem2) {
  int tid = blockIdx.x * blockDim.x + threadIdx.x;
  if (tid >= E_EDGES * K_NBR * M_DIM) return;
  int m = tid % M_DIM;
  int ej = tid / M_DIM;
  int e = ej / K_NBR, j = ej - e * K_NBR;
  float il = is_last[e];
  if (il == 0.0f) return;
  int node = nodes[e];
  int nbr = neighbors[(size_t)node * K_NBR + j];
  float dt = times[e] - last_upd[nbr];
  dt = fminf(fmaxf(dt, 0.0f), 50.0f);
  float decay = expf(-dt / TAUF);
  unsafeAtomicAdd(&mem2[(size_t)nbr * M_DIM + m], il * decay * prop[(size_t)e * M_DIM + m]);
}

// ---------- row-dot + sigmoid ----------
__global__ void dot_sigmoid_kernel(const float* __restrict__ u, const float* __restrict__ v,
                                   float* __restrict__ out) {
  int e = blockIdx.x * blockDim.x + threadIdx.x;
  if (e >= E_EDGES) return;
  float s = 0.0f;
  for (int i = 0; i < DM_DIM; ++i)
    s += u[(size_t)e * DM_DIM + i] * v[(size_t)e * DM_DIM + i];
  out[e] = sigmoidf_(s);
}

extern "C" void kernel_launch(void* const* d_in, const int* in_sizes, int n_in,
                              void* d_out, int out_size, void* d_ws, size_t ws_size,
                              hipStream_t stream) {
  (void)in_sizes; (void)n_in; (void)out_size; (void)ws_size;
  const float* memory_s  = (const float*)d_in[0];
  const float* memory_g  = (const float*)d_in[1];
  const float* last_up_s = (const float*)d_in[2];
  const float* last_up_g = (const float*)d_in[3];
  const float* edge_feats= (const float*)d_in[4];
  const float* edge_times= (const float*)d_in[5];
  const int*   src       = (const int*)d_in[6];
  const int*   dst       = (const int*)d_in[7];
  const int*   neg       = (const int*)d_in[8];
  const int*   eidx      = (const int*)d_in[9];
  const int*   neighbors = (const int*)d_in[10];
  const float* W_merge   = (const float*)d_in[11];
  const float* b_merge   = (const float*)d_in[12];
  const float* W_msg     = (const float*)d_in[13];
  const float* b_msg     = (const float*)d_in[14];
  const float* Wx_s      = (const float*)d_in[15];
  const float* Wh_s      = (const float*)d_in[16];
  const float* bg_s      = (const float*)d_in[17];
  const float* Wx_g      = (const float*)d_in[18];
  const float* Wh_g      = (const float*)d_in[19];
  const float* bg_g      = (const float*)d_in[20];
  const float* Wp_s      = (const float*)d_in[21];
  const float* Wp_g      = (const float*)d_in[22];
  const float* W_s       = (const float*)d_in[23];
  const float* W_g       = (const float*)d_in[24];
  float* out = (float*)d_out;

  // ---- workspace carve-out ----
  char* p = (char*)d_ws;
  auto alloc = [&](size_t bytes) -> void* {
    void* r = (void*)p;
    p += (bytes + 255) & ~(size_t)255;
    return r;
  };
  float* mem_s2 = (float*)alloc((size_t)N_NODES * M_DIM * 4);
  float* mem_g2 = (float*)alloc((size_t)N_NODES * M_DIM * 4);
  float* m_a    = (float*)alloc((size_t)E_EDGES * M_DIM * 4);   // src merges
  float* m_b    = (float*)alloc((size_t)E_EDGES * M_DIM * 4);   // dst merges
  float* m_c    = (float*)alloc((size_t)E_EDGES * M_DIM * 4);   // neg merges
  float* msg_s  = (float*)alloc((size_t)E_EDGES * DM_DIM * 4);
  float* msg_g  = (float*)alloc((size_t)E_EDGES * DM_DIM * 4);
  float* gx     = (float*)alloc((size_t)E_EDGES * G3M * 4);
  float* gh     = (float*)alloc((size_t)E_EDGES * G3M * 4);
  float* prop   = (float*)alloc((size_t)E_EDGES * M_DIM * 4);
  float* u      = (float*)alloc((size_t)E_EDGES * DM_DIM * 4);
  float* vv     = (float*)alloc((size_t)E_EDGES * DM_DIM * 4);
  float* il_s   = (float*)alloc((size_t)E_EDGES * 4);
  float* il_g   = (float*)alloc((size_t)E_EDGES * 4);
  int*   lo_s   = (int*)alloc((size_t)N_NODES * 4);
  int*   lo_g   = (int*)alloc((size_t)N_NODES * 4);
  __bf16* Wm_sw   = (__bf16*)alloc((size_t)13 * 13 * 512 * 2);
  __bf16* Wmsg_sw = (__bf16*)alloc((size_t)18 * 7 * 512 * 2);
  __bf16* Wxs_sw  = (__bf16*)alloc((size_t)4 * 38 * 512 * 2);
  __bf16* Whs_sw  = (__bf16*)alloc((size_t)7 * 38 * 512 * 2);
  __bf16* Wps_sw  = (__bf16*)alloc((size_t)4 * 13 * 512 * 2);
  __bf16* Wxg_sw  = (__bf16*)alloc((size_t)4 * 38 * 512 * 2);
  __bf16* Whg_sw  = (__bf16*)alloc((size_t)7 * 38 * 512 * 2);
  __bf16* Wpg_sw  = (__bf16*)alloc((size_t)4 * 13 * 512 * 2);
  __bf16* Ws_sw   = (__bf16*)alloc((size_t)7 * 7 * 512 * 2);
  __bf16* Wg_sw   = (__bf16*)alloc((size_t)7 * 7 * 512 * 2);

  // ---- materialize updated memories (D2D copies are graph-capturable) ----
  hipMemcpyAsync(mem_s2, memory_s, (size_t)N_NODES * M_DIM * 4, hipMemcpyDeviceToDevice, stream);
  hipMemcpyAsync(mem_g2, memory_g, (size_t)N_NODES * M_DIM * 4, hipMemcpyDeviceToDevice, stream);

  // ---- last-message flags ----
  init_lastocc_kernel<<<(N_NODES + 255) / 256, 256, 0, stream>>>(lo_s, N_NODES);
  init_lastocc_kernel<<<(N_NODES + 255) / 256, 256, 0, stream>>>(lo_g, N_NODES);
  scatter_last_kernel<<<(E_EDGES + 255) / 256, 256, 0, stream>>>(src, dst, lo_s, lo_g);
  is_last_kernel<<<(E_EDGES + 255) / 256, 256, 0, stream>>>(src, dst, lo_s, lo_g, il_s, il_g);

  // ---- pack weights to bf16 fragment layout ----
  auto pack = [&](const float* W, __bf16* o, int K, int Nc, int nKt, int nCt) {
    int total = nKt * nCt * 512;
    pack_weight_kernel<<<(total + 255) / 256, 256, 0, stream>>>(W, o, K, Nc, nKt, nCt);
  };
  pack(W_merge, Wm_sw, 400, 200, 13, 13);
  pack(W_msg, Wmsg_sw, 572, 100, 18, 7);
  pack(Wx_s, Wxs_sw, 100, 600, 4, 38);
  pack(Wh_s, Whs_sw, 200, 600, 7, 38);
  pack(Wp_s, Wps_sw, 100, 200, 4, 13);
  pack(Wx_g, Wxg_sw, 100, 600, 4, 38);
  pack(Wh_g, Whg_sw, 200, 600, 7, 38);
  pack(Wp_g, Wpg_sw, 100, 200, 4, 13);
  pack(W_s, Ws_sw, 200, 100, 7, 7);
  pack(W_g, Wg_sw, 200, 100, 7, 7);

  const dim3 blk(128);
  const int RT = E_EDGES / 16;  // 512 row tiles

  // ---- pre-update merges + messages ----
  merge_gemm_kernel<<<dim3(RT, 4), blk, 0, stream>>>(memory_s, memory_g, src, Wm_sw, b_merge, m_a);
  merge_gemm_kernel<<<dim3(RT, 4), blk, 0, stream>>>(memory_s, memory_g, dst, Wm_sw, b_merge, m_b);
  msg_gemm_kernel<<<dim3(RT, 2), blk, 0, stream>>>(m_a, m_b, edge_feats, eidx, Wmsg_sw, b_msg, msg_s);
  msg_gemm_kernel<<<dim3(RT, 2), blk, 0, stream>>>(m_b, m_a, edge_feats, eidx, Wmsg_sw, b_msg, msg_g);

  // ---- side s: GRU + propagation into mem_s2 ----
  dense_gemm_kernel<4, 38><<<dim3(RT, 10), blk, 0, stream>>>(msg_s, nullptr, DM_DIM, DM_DIM, Wxs_sw, bg_s, gx, G3M, 0);
  dense_gemm_kernel<7, 38><<<dim3(RT, 10), blk, 0, stream>>>(memory_s, src, M_DIM, M_DIM, Whs_sw, nullptr, gh, G3M, 0);
  dense_gemm_kernel<4, 13><<<dim3(RT, 4), blk, 0, stream>>>(msg_s, nullptr, DM_DIM, DM_DIM, Wps_sw, nullptr, prop, M_DIM, 1);
  gru_scatter_kernel<<<(E_EDGES * M_DIM + 255) / 256, 256, 0, stream>>>(gx, gh, memory_s, src, il_s, mem_s2);
  prop_scatter_kernel<<<(E_EDGES * K_NBR * M_DIM + 255) / 256, 256, 0, stream>>>(prop, src, neighbors, edge_times, last_up_s, il_s, mem_s2);

  // ---- side g ----
  dense_gemm_kernel<4, 38><<<dim3(RT, 10), blk, 0, stream>>>(msg_g, nullptr, DM_DIM, DM_DIM, Wxg_sw, bg_g, gx, G3M, 0);
  dense_gemm_kernel<7, 38><<<dim3(RT, 10), blk, 0, stream>>>(memory_g, dst, M_DIM, M_DIM, Whg_sw, nullptr, gh, G3M, 0);
  dense_gemm_kernel<4, 13><<<dim3(RT, 4), blk, 0, stream>>>(msg_g, nullptr, DM_DIM, DM_DIM, Wpg_sw, nullptr, prop, M_DIM, 1);
  gru_scatter_kernel<<<(E_EDGES * M_DIM + 255) / 256, 256, 0, stream>>>(gx, gh, memory_g, dst, il_g, mem_g2);
  prop_scatter_kernel<<<(E_EDGES * K_NBR * M_DIM + 255) / 256, 256, 0, stream>>>(prop, dst, neighbors, edge_times, last_up_g, il_g, mem_g2);

  // ---- scores from updated memories ----
  merge_gemm_kernel<<<dim3(RT, 4), blk, 0, stream>>>(mem_s2, mem_g2, src, Wm_sw, b_merge, m_a);
  merge_gemm_kernel<<<dim3(RT, 4), blk, 0, stream>>>(mem_s2, mem_g2, dst, Wm_sw, b_merge, m_b);
  merge_gemm_kernel<<<dim3(RT, 4), blk, 0, stream>>>(mem_s2, mem_g2, neg, Wm_sw, b_merge, m_c);
  dense_gemm_kernel<7, 7><<<dim3(RT, 2), blk, 0, stream>>>(m_a, nullptr, M_DIM, M_DIM, Ws_sw, nullptr, u, DM_DIM, 0);
  dense_gemm_kernel<7, 7><<<dim3(RT, 2), blk, 0, stream>>>(m_b, nullptr, M_DIM, M_DIM, Wg_sw, nullptr, vv, DM_DIM, 0);
  dot_sigmoid_kernel<<<(E_EDGES + 255) / 256, 256, 0, stream>>>(u, vv, out);
  dense_gemm_kernel<7, 7><<<dim3(RT, 2), blk, 0, stream>>>(m_c, nullptr, M_DIM, M_DIM, Wg_sw, nullptr, vv, DM_DIM, 0);
  dot_sigmoid_kernel<<<(E_EDGES + 255) / 256, 256, 0, stream>>>(u, vv, out + E_EDGES);
}
